// RBM_81801947120402
// MI455X (gfx1250) — matrix operand order repaired and checked by reading, not compiled, not run on hardware
//
#include <hip/hip_runtime.h>
#include <hip/hip_bf16.h>

// ---------------------------------------------------------------------------
// RBM CD-10 fused kernel for MI455X (gfx1250, wave32, WMMA).
//
// Roofline: noise tensors (671 MB, read once) bound the kernel at ~29us @
// 23.3 TB/s; the 49 GFLOP of GEMM work hides behind that via
// v_wmma_f32_16x16x32_f16. W (fp16, both layouts) lives in LDS.
// This revision: fast-rcp sigmoid (4 VALU/elem instead of IEEE divide) and
// batched b128 noise loads (pipelined) instead of serialized dword loads.
// ---------------------------------------------------------------------------

typedef __attribute__((ext_vector_type(16))) _Float16 v16h;
typedef __attribute__((ext_vector_type(8)))  _Float16 v8h;
typedef __attribute__((ext_vector_type(4)))  _Float16 v4h;
typedef __attribute__((ext_vector_type(8)))  float    v8f;
typedef __attribute__((ext_vector_type(4)))  float    v4f;

#define BTOT  65536
#define NV    128
#define NH    128
#define KST   10
#define ROWS  128          // batch rows per block (8 waves x 16)
#define L2REG 1.0e-4f
#define L1REG 0.0f

// sigmoid via v_mul + v_exp + v_add + v_rcp (no IEEE-divide expansion)
__device__ __forceinline__ float sigmoidf(float x) {
    float e = __builtin_amdgcn_exp2f(x * -1.4426950408889634f);  // exp(-x)
    return __builtin_amdgcn_rcpf(1.0f + e);
}

// A fragment (16x32 f16, M=16 rows, K=32): row-major source, row stride 128.
// Lane L: m = L&15, 16 contiguous K at (L>>4)*16 (matches load_bt).
__device__ __forceinline__ v16h load_a(const _Float16* base, int lane) {
    const int m  = lane & 15;
    const int hf = lane >> 4;
    const _Float16* p = base + m * 128 + hf * 16;
    v8h lo = *(const v8h*)(p);
    v8h hi = *(const v8h*)(p + 8);
    return __builtin_shufflevector(lo, hi, 0,1,2,3,4,5,6,7,8,9,10,11,12,13,14,15);
}

// B fragment (32x16 f16) from a "Bt" buffer laid out Bt[n][k] (row stride
// 128): lane L -> n = L&15, 16 contiguous K at (L>>4)*16.
__device__ __forceinline__ v16h load_bt(const _Float16* base, int lane) {
    const int n  = lane & 15;
    const int kg = lane >> 4;
    return *(const v16h*)(base + n * 128 + kg * 16);
}

__device__ __forceinline__ v8f wmma_f16(v16h a, v16h b, v8f c) {
    return __builtin_amdgcn_wmma_f32_16x16x32_f16(false, a, false, b,
                                                  (short)0, c, false, false);
}

// In-place Bernoulli sampling of a wave-private 16x128 fp16 tile that holds
// probabilities: tile[e] = (noise[e] < tile[e]) ? 1 : 0.  Noise has the same
// [row][col] linear layout, so lane l / chunk j is a contiguous float4.
// All 16 b128 noise loads are issued before any use (pipelined).
__device__ __forceinline__ void sample_inplace(_Float16* tile,
                                               const float* noise, int lane) {
    v4f nz[16];
    #pragma unroll
    for (int j = 0; j < 16; j++)
        nz[j] = *(const v4f*)(noise + lane * 4 + j * 128);
    #pragma unroll
    for (int j = 0; j < 16; j++) {
        _Float16* p = tile + lane * 4 + j * 128;
        v4h ph = *(const v4h*)p;
        v4h o;
        #pragma unroll
        for (int i = 0; i < 4; i++)
            o[i] = (nz[j][i] < (float)ph[i]) ? (_Float16)1.0f : (_Float16)0.0f;
        *(v4h*)p = o;
    }
}

// Prefetch one wave-slice (8KB) of noise: each lane touches a distinct 256B line.
__device__ __forceinline__ void prefetch_noise(const float* noise, int lane) {
    __builtin_prefetch(noise + lane * 64, 0, 1);   // -> global_prefetch_b8
}

extern "C" __global__ void __launch_bounds__(256, 1)
rbm_main(const float* __restrict__ batch,
         const float* __restrict__ W,
         const float* __restrict__ vbias,
         const float* __restrict__ hbias,
         const float* __restrict__ nh_all,   // [K,B,H]
         const float* __restrict__ nv_all,   // [K,B,V]
         float* __restrict__ gw,             // [H,V] accumulated (raw sums)
         float* __restrict__ gvb,            // [V]
         float* __restrict__ ghb)            // [H]
{
    extern __shared__ __attribute__((aligned(32))) char smem[];
    _Float16* Wrm  = (_Float16*)smem;        // [h][v]  = W           (32KB)
    _Float16* Wcm  = Wrm  + 16384;           // [v][h]  = W^T         (32KB)
    _Float16* vbuf = Wcm  + 16384;           // [row][v] current v    (32KB)
    _Float16* hbuf = vbuf + 16384;           // [row][h] current h    (32KB)
    _Float16* vt   = hbuf + 16384;           // [v][row] batch_t/vk_t (32KB)
    _Float16* pht  = vt   + 16384;           // [h][row] ph0_t/phk_t  (32KB)
    float* hbL  = (float*)(pht + 16384);     // 128
    float* vbL  = hbL + 128;                 // 128
    float* hsum = vbL + 128;                 // 128
    float* vsum = hsum + 128;                // 128

    const int tid  = threadIdx.x;
    const int wave = tid >> 5;
    const int lane = tid & 31;
    const int hf   = lane >> 4;
    const int lq   = lane & 15;
    const int m0   = wave * 16;              // block-local row base of this wave
    const size_t row0 = (size_t)blockIdx.x * ROWS;

    if (tid < 128) {
        hbL[tid] = hbias[tid];
        vbL[tid] = vbias[tid];
        hsum[tid] = 0.0f;
        vsum[tid] = 0.0f;
    }
    __syncthreads();

    // Load W into LDS in both layouts (fp16).
    for (int i = tid; i < 16384; i += 256) {
        float w = W[i];
        int h = i >> 7, v = i & 127;
        _Float16 wh = (_Float16)w;
        Wrm[i] = wh;
        Wcm[v * 128 + h] = wh;
    }
    // Load batch tile (fp16, both layouts) and accumulate -sum(batch) per col.
    float vs_local = 0.0f;
    for (int i = tid; i < ROWS * 128; i += 256) {
        int r = i >> 7, c = i & 127;         // c == tid&127 (stride-256, width-128)
        float bv = batch[(row0 + r) * 128 + c];
        _Float16 bh = (_Float16)bv;
        vbuf[r * 128 + c] = bh;
        vt[c * 128 + r]   = bh;
        vs_local -= bv;
    }
    atomicAdd(&vsum[tid & 127], vs_local);
    __syncthreads();

    v8f   accW[8];                           // wave's 16 h-rows x 128 v-cols
    float hs[8];                             // per-lane h-column sums (phk-ph0)
    #pragma unroll
    for (int t = 0; t < 8; t++) {
        #pragma unroll
        for (int r = 0; r < 8; r++) accW[t][r] = 0.0f;
        hs[t] = 0.0f;
    }

    // ------------------------- CD-k Gibbs chain -------------------------
    for (int k = 0; k < KST; k++) {
        const float* nh_k = nh_all + ((size_t)k * BTOT + row0 + m0) * NH;
        const float* nv_k = nv_all + ((size_t)k * BTOT + row0 + m0) * NV;
        prefetch_noise(nv_k, lane);          // v-step noise needed shortly

        // h step: p = sigmoid(v @ W^T + hb) -> hbuf (probabilities)
        v16h a0 = load_a(vbuf + m0 * 128 +  0, lane);
        v16h a1 = load_a(vbuf + m0 * 128 + 32, lane);
        v16h a2 = load_a(vbuf + m0 * 128 + 64, lane);
        v16h a3 = load_a(vbuf + m0 * 128 + 96, lane);
        #pragma unroll
        for (int t = 0; t < 8; t++) {
            v8f acc = {};
            acc = wmma_f16(a0, load_bt(Wrm + t * 16 * 128 +  0, lane), acc);
            acc = wmma_f16(a1, load_bt(Wrm + t * 16 * 128 + 32, lane), acc);
            acc = wmma_f16(a2, load_bt(Wrm + t * 16 * 128 + 64, lane), acc);
            acc = wmma_f16(a3, load_bt(Wrm + t * 16 * 128 + 96, lane), acc);
            const int col = t * 16 + lq;
            const float bias = hbL[col];
            #pragma unroll
            for (int r = 0; r < 8; r++) {
                const int m = r + 8 * hf;    // row within wave tile
                float p = sigmoidf(acc[r] + bias);
                if (k == 0) {                // ph0 for positive phase
                    pht[col * 128 + m0 + m] = (_Float16)p;
                    hs[t] -= p;
                }
                hbuf[(m0 + m) * 128 + col] = (_Float16)p;
            }
        }
        // sample h in place: h = (noise_h < p)
        sample_inplace(hbuf + m0 * 128, nh_k, lane);

        if (k == 0) {
            // Positive phase: accW -= ph0^T @ batch (K = 128 block rows).
            __syncthreads();                 // pht written by all waves
            v16h p0 = load_a(pht + m0 * 128 +  0, lane);
            v16h p1 = load_a(pht + m0 * 128 + 32, lane);
            v16h p2 = load_a(pht + m0 * 128 + 64, lane);
            v16h p3 = load_a(pht + m0 * 128 + 96, lane);
            #pragma unroll
            for (int t = 0; t < 8; t++) {
                v8f c = {};
                c = wmma_f16(p0, load_bt(vt + t * 16 * 128 +  0, lane), c);
                c = wmma_f16(p1, load_bt(vt + t * 16 * 128 + 32, lane), c);
                c = wmma_f16(p2, load_bt(vt + t * 16 * 128 + 64, lane), c);
                c = wmma_f16(p3, load_bt(vt + t * 16 * 128 + 96, lane), c);
                #pragma unroll
                for (int r = 0; r < 8; r++) accW[t][r] -= c[r];
            }
        }

        if (k + 1 < KST)                     // pre-touch next h-step noise
            prefetch_noise(nh_all + ((size_t)(k + 1) * BTOT + row0 + m0) * NH, lane);

        // v step: q = sigmoid(h @ W + vb) -> vbuf (probabilities)
        v16h h0 = load_a(hbuf + m0 * 128 +  0, lane);
        v16h h1 = load_a(hbuf + m0 * 128 + 32, lane);
        v16h h2 = load_a(hbuf + m0 * 128 + 64, lane);
        v16h h3 = load_a(hbuf + m0 * 128 + 96, lane);
        #pragma unroll
        for (int t = 0; t < 8; t++) {
            v8f acc = {};
            acc = wmma_f16(h0, load_bt(Wcm + t * 16 * 128 +  0, lane), acc);
            acc = wmma_f16(h1, load_bt(Wcm + t * 16 * 128 + 32, lane), acc);
            acc = wmma_f16(h2, load_bt(Wcm + t * 16 * 128 + 64, lane), acc);
            acc = wmma_f16(h3, load_bt(Wcm + t * 16 * 128 + 96, lane), acc);
            const int col = t * 16 + lq;
            const float bias = vbL[col];
            #pragma unroll
            for (int r = 0; r < 8; r++) {
                const int m = r + 8 * hf;
                float q = sigmoidf(acc[r] + bias);
                vbuf[(m0 + m) * 128 + col] = (_Float16)q;
            }
        }
        // sample v in place: v = (noise_v < q)
        sample_inplace(vbuf + m0 * 128, nv_k, lane);
    }

    // ----------------------- Negative (model) phase -----------------------
    __syncthreads();
    // Transpose vk into vt and accumulate +sum(vk) per visible column.
    float vs2 = 0.0f;
    for (int i = tid; i < ROWS * 128; i += 256) {
        int r = i >> 7, c = i & 127;
        _Float16 x = vbuf[r * 128 + c];
        vt[c * 128 + r] = x;
        vs2 += (float)x;
    }
    atomicAdd(&vsum[tid & 127], vs2);
    __syncthreads();

    // phk = sigmoid(vk @ W^T + hb) -> pht ; hs += phk
    {
        v16h a0 = load_a(vbuf + m0 * 128 +  0, lane);
        v16h a1 = load_a(vbuf + m0 * 128 + 32, lane);
        v16h a2 = load_a(vbuf + m0 * 128 + 64, lane);
        v16h a3 = load_a(vbuf + m0 * 128 + 96, lane);
        #pragma unroll
        for (int t = 0; t < 8; t++) {
            v8f acc = {};
            acc = wmma_f16(a0, load_bt(Wrm + t * 16 * 128 +  0, lane), acc);
            acc = wmma_f16(a1, load_bt(Wrm + t * 16 * 128 + 32, lane), acc);
            acc = wmma_f16(a2, load_bt(Wrm + t * 16 * 128 + 64, lane), acc);
            acc = wmma_f16(a3, load_bt(Wrm + t * 16 * 128 + 96, lane), acc);
            const int col = t * 16 + lq;
            const float bias = hbL[col];
            #pragma unroll
            for (int r = 0; r < 8; r++) {
                const int m = r + 8 * hf;
                float p = sigmoidf(acc[r] + bias);
                pht[col * 128 + m0 + m] = (_Float16)p;
                hs[t] += p;
            }
        }
    }
    __syncthreads();

    // accW += phk^T @ vk
    {
        v16h p0 = load_a(pht + m0 * 128 +  0, lane);
        v16h p1 = load_a(pht + m0 * 128 + 32, lane);
        v16h p2 = load_a(pht + m0 * 128 + 64, lane);
        v16h p3 = load_a(pht + m0 * 128 + 96, lane);
        #pragma unroll
        for (int t = 0; t < 8; t++) {
            v8f c = {};
            c = wmma_f16(p0, load_bt(vt + t * 16 * 128 +  0, lane), c);
            c = wmma_f16(p1, load_bt(vt + t * 16 * 128 + 32, lane), c);
            c = wmma_f16(p2, load_bt(vt + t * 16 * 128 + 64, lane), c);
            c = wmma_f16(p3, load_bt(vt + t * 16 * 128 + 96, lane), c);
            #pragma unroll
            for (int r = 0; r < 8; r++) accW[t][r] += c[r];
        }
    }

    // ------------------------- Flush accumulators -------------------------
    #pragma unroll
    for (int t = 0; t < 8; t++) atomicAdd(&hsum[t * 16 + lq], hs[t]);

    #pragma unroll
    for (int t = 0; t < 8; t++) {
        const int vcol = t * 16 + lq;
        #pragma unroll
        for (int r = 0; r < 8; r++) {
            const int hrow = m0 + r + 8 * hf;
            atomicAdd(&gw[hrow * 128 + vcol], accW[t][r]);
        }
    }
    __syncthreads();
    if (tid < 128) {
        atomicAdd(&gvb[tid], vsum[tid]);
        atomicAdd(&ghb[tid], hsum[tid]);
    }
}

extern "C" __global__ void rbm_zero(float* __restrict__ out, int n) {
    int i = blockIdx.x * 256 + threadIdx.x;
    if (i < n) out[i] = 0.0f;
}

// Scale raw sums by 1/B and add regularization; emit phase gradients.
extern "C" __global__ void rbm_final(float* __restrict__ out,
                                     const float* __restrict__ W,
                                     const float* __restrict__ Wp) {
    int i = blockIdx.x * 256 + threadIdx.x;
    const float inv = 1.0f / (float)BTOT;
    if (i < 16384) {
        float w  = W[i];
        float s  = (w > 0.0f) ? 1.0f : ((w < 0.0f) ? -1.0f : 0.0f);
        out[i] = out[i] * inv + L2REG * w + L1REG * s;
        float wp = Wp[i];
        float sp = (wp > 0.0f) ? 1.0f : ((wp < 0.0f) ? -1.0f : 0.0f);
        out[16640 + i] = L2REG * wp + L1REG * sp;     // g_w_phase
    } else if (i < 16640) {
        out[i] *= inv;                                // g_vb_amp / g_hb_amp
    }
    // phase biases stay zero (rbm_zero already wrote them)
}

extern "C" void kernel_launch(void* const* d_in, const int* in_sizes, int n_in,
                              void* d_out, int out_size, void* d_ws, size_t ws_size,
                              hipStream_t stream) {
    const float* batch = (const float*)d_in[0];
    const float* W     = (const float*)d_in[1];
    const float* vb    = (const float*)d_in[2];
    const float* hb    = (const float*)d_in[3];
    const float* Wp    = (const float*)d_in[4];
    // d_in[5] = visible_bias_phase, d_in[6] = hidden_bias_phase (unused: grads are reg-only/zero)
    const float* nh    = (const float*)d_in[7];
    const float* nv    = (const float*)d_in[8];
    float* out = (float*)d_out;

    const int total_out = 16384 + 128 + 128 + 16384 + 128 + 128;   // 33280
    rbm_zero<<<(total_out + 255) / 256, 256, 0, stream>>>(out, total_out);

    const size_t lds_bytes = 6 * 16384 * sizeof(_Float16) + 4 * 128 * sizeof(float);
    rbm_main<<<BTOT / ROWS, 256, lds_bytes, stream>>>(
        batch, W, vb, hb, nh, nv,
        out,                 // g_w_amp  [H,V]
        out + 16384,         // g_vb_amp [V]
        out + 16512);        // g_hb_amp [H]

    rbm_final<<<(16640 + 255) / 256, 256, 0, stream>>>(out, W, Wp);
}
